// Gunn2D_1494648619348
// MI455X (gfx1250) — matrix operand order repaired and checked by reading, not compiled, not run on hardware
//
#include <hip/hip_runtime.h>

#define CH   256
#define EX   64
#define NPIX 65536      // 16*64*64
#define HW   64
#define TW   16
#define TH   4
#define KC   128        // K chunk staged in LDS for the 3x3 conv

#if __has_builtin(__builtin_amdgcn_global_load_async_to_lds_b128)
#define ASYNC_LDS 1
#endif

typedef __attribute__((ext_vector_type(8)))  float  v8f;
typedef __attribute__((ext_vector_type(16))) __bf16 v16bf;
typedef int v4i __attribute__((vector_size(16)));

union Frag16 {
  uint4  q[2];
  v16bf  v;
};

union U4H8 {
  uint4          q;
  unsigned short h[8];
};

__device__ __forceinline__ unsigned short f2bf(float f) {
  unsigned int u = __float_as_uint(f);
  u += 0x7FFFu + ((u >> 16) & 1u);
  return (unsigned short)(u >> 16);
}

__device__ __forceinline__ unsigned int pack2bf(float a, float b) {
  return (unsigned int)f2bf(a) | ((unsigned int)f2bf(b) << 16);
}

__device__ __forceinline__ float bf2f(unsigned short h) {
  return __uint_as_float((unsigned int)h << 16);
}

// ---- gfx1250 async global->LDS copy (ASYNCcnt-tracked), 16B per lane -------
__device__ __forceinline__ void async_g2l_b128(const void* gsrc, void* ldst) {
#ifdef ASYNC_LDS
  __builtin_amdgcn_global_load_async_to_lds_b128(
      (__attribute__((address_space(1))) v4i*)(unsigned long long)gsrc,
      (__attribute__((address_space(3))) v4i*)(unsigned int)(unsigned long long)ldst,
      0, 0);
#else
  *(uint4*)ldst = *(const uint4*)gsrc;
#endif
}

__device__ __forceinline__ void async_wait0() {
#ifdef ASYNC_LDS
#if __has_builtin(__builtin_amdgcn_s_wait_asynccnt)
  __builtin_amdgcn_s_wait_asynccnt(0);
#else
  asm volatile("s_wait_asynccnt 0" ::: "memory");
#endif
#endif
}

// ---------------------------------------------------------------------------
// Repack weights: w1 (1,1,256,64)->bf16 [o][i]; w2 (3,3,256,64)->bf16 [tap][o][i]
// Also zero-fill the 1 KiB zero page used for SAME-padding async reads.
// ---------------------------------------------------------------------------
__global__ __launch_bounds__(256) void repack_kernel(
    const float* __restrict__ w1, const float* __restrict__ w2,
    unsigned short* __restrict__ w1T, unsigned short* __restrict__ w2T,
    unsigned short* __restrict__ zeropad)
{
  const int nt = gridDim.x * blockDim.x;
  if (blockIdx.x == 0 && threadIdx.x < 64)
    ((uint4*)zeropad)[threadIdx.x] = make_uint4(0u, 0u, 0u, 0u);
  for (int t = blockIdx.x * blockDim.x + threadIdx.x; t < EX * CH; t += nt) {
    const int o = t >> 8, i = t & 255;
    w1T[t] = f2bf(w1[i * EX + o]);
  }
  for (int t = blockIdx.x * blockDim.x + threadIdx.x; t < 9 * EX * CH; t += nt) {
    const int tap = t / (EX * CH);
    const int r   = t - tap * (EX * CH);
    const int o = r >> 8, i = r & 255;
    w2T[t] = f2bf(w2[(tap * CH + i) * EX + o]);
  }
}

// ---------------------------------------------------------------------------
// step1: 4 gradual 1x1 GEMMs per pixel tile + ReLU -> A (bf16 master)
// Block = 128 thr = 4 waves; wave = 16 pixels x 64 outputs via WMMA bf16
// ---------------------------------------------------------------------------
__global__ __launch_bounds__(128) void step1_kernel(
    const float* __restrict__ x, const unsigned short* __restrict__ w1T,
    const float* __restrict__ b1, unsigned short* __restrict__ Abf)
{
  __shared__ alignas(16) unsigned short ldsA[64 * CH];   // 32 KiB, bf16 tile
  const int tid  = threadIdx.x;
  const int lane = tid & 31;
  const int wv   = tid >> 5;
  const long pixBase = (long)blockIdx.x * 64;

  // stage x tile -> bf16 LDS (contiguous 64*256 floats; fp32->bf16 in VALU)
  {
    const float4* src = (const float4*)(x + pixBase * CH);
    for (int idx = tid; idx < 64 * CH / 4; idx += 128) {
      float4 f = src[idx];
      uint2 pk;
      pk.x = pack2bf(f.x, f.y);
      pk.y = pack2bf(f.z, f.w);
      *(uint2*)&ldsA[idx * 4] = pk;
    }
  }
  __syncthreads();

  const int row0 = wv * 16;
  const int mrow = lane & 15;
  const int hi   = lane >> 4;

  float bias[4];
#pragma unroll
  for (int n = 0; n < 4; ++n) bias[n] = b1[n * 16 + mrow];

  const unsigned short* arow = &ldsA[(row0 + mrow) * CH];

  for (int i = 0; i < 4; ++i) {
    v8f acc[4] = {};
#pragma unroll
    for (int k0 = 0; k0 < CH; k0 += 32) {
      Frag16 a;
      a.q[0] = *(const uint4*)&arow[k0 + hi * 16];
      a.q[1] = *(const uint4*)&arow[k0 + hi * 16 + 8];
#pragma unroll
      for (int n = 0; n < 4; ++n) {
        Frag16 bfr;
        const uint4* bw = (const uint4*)(w1T + (n * 16 + mrow) * CH + k0 + hi * 16);
        bfr.q[0] = bw[0];
        bfr.q[1] = bw[1];
        acc[n] = __builtin_amdgcn_wmma_f32_16x16x32_bf16(
            false, a.v, false, bfr.v, (short)0, acc[n], false, false);
      }
    }
    // overwrite slice i: LDS bf16 copy feeds the next gradual GEMM (pre-ReLU),
    // ReLU'd bf16 goes to the global master A (post-step1 semantics).
#pragma unroll
    for (int n = 0; n < 4; ++n) {
      const int col = i * EX + n * 16 + mrow;
#pragma unroll
      for (int e = 0; e < 8; ++e) {
        const float z   = acc[n][e] + bias[n];
        const int   row = row0 + e + hi * 8;
        ldsA[row * CH + col] = f2bf(z);
        Abf[(pixBase + row) * CH + col] = f2bf(fmaxf(z, 0.f));
      }
    }
  }
}

// ---------------------------------------------------------------------------
// step2: one 3x3 SAME conv (256->64) reading bf16 A, writing compact Z (fp32).
// Launched 4x, commit kernel merges the slice (stream order = grid sync).
// Tile 16x4 pixels, halo 18x6, K staged in 128-channel LDS chunks via the
// gfx1250 async global->LDS DMA path (ASYNCcnt). OOB halo reads a zero page.
// ---------------------------------------------------------------------------
__global__ __launch_bounds__(128) void step2_kernel(
    const unsigned short* __restrict__ Abf, const unsigned short* __restrict__ w2T,
    const float* __restrict__ b2, float* __restrict__ Z,
    const unsigned short* __restrict__ zeropad)
{
  __shared__ alignas(16) unsigned short ldsA[(TH + 2) * (TW + 2) * KC]; // 27 KiB
  __shared__ alignas(16) unsigned short ldsW[EX * KC];                  // 16 KiB

  const int tid  = threadIdx.x;
  const int lane = tid & 31;
  const int wv   = tid >> 5;
  const int x0   = blockIdx.x * TW;
  const int y0   = blockIdx.y * TH;
  const long imgBase = (long)blockIdx.z * (HW * HW);

  const int mrow = lane & 15;
  const int hi   = lane >> 4;

  v8f acc[4] = {};

  for (int kc = 0; kc < CH; kc += KC) {
    __syncthreads();  // everyone done reading ldsA from the previous chunk
    // stage halo tile chunk: 108 pixels x 128 channels, 16 lanes/pixel x 16 B
    {
      const int seg = tid & 15;
      for (int p = tid >> 4; p < (TH + 2) * (TW + 2); p += 8) {
        const int hy = p / (TW + 2);
        const int hx = p - hy * (TW + 2);
        const int gy = y0 + hy - 1;
        const int gx = x0 + hx - 1;
        const unsigned short* gsrc =
            (gy >= 0 && gy < HW && gx >= 0 && gx < HW)
                ? Abf + (imgBase + (long)gy * HW + gx) * CH + kc + seg * 8
                : zeropad + seg * 8;
        async_g2l_b128(gsrc, &ldsA[p * KC + seg * 8]);
      }
    }

    for (int tap = 0; tap < 9; ++tap) {
      __syncthreads();  // everyone done reading ldsW from the previous tap
      // stage this tap's weight chunk [64 out][128 in] bf16 (async DMA)
      for (int idx = tid; idx < EX * (KC / 8); idx += 128) {
        const int o   = idx >> 4;        // output row
        const int c8  = idx & 15;        // 16B segment within the chunk
        async_g2l_b128(w2T + (tap * EX + o) * CH + kc + c8 * 8,
                       &ldsW[idx * 8]);
      }
      async_wait0();     // our ASYNCcnt drained (covers A-chunk on tap 0 too)
      __syncthreads();   // all waves' DMA landed in LDS

      const int dy = tap / 3, dx = tap - dy * 3;
      const unsigned short* arow =
          &ldsA[((wv + dy) * (TW + 2) + (mrow + dx)) * KC];
#pragma unroll
      for (int k0 = 0; k0 < KC; k0 += 32) {
        Frag16 a;
        a.q[0] = *(const uint4*)&arow[k0 + hi * 16];
        a.q[1] = *(const uint4*)&arow[k0 + hi * 16 + 8];
#pragma unroll
        for (int n = 0; n < 4; ++n) {
          Frag16 bfr;
          const uint4* bw = (const uint4*)&ldsW[(n * 16 + mrow) * KC + k0 + hi * 16];
          bfr.q[0] = bw[0];
          bfr.q[1] = bw[1];
          acc[n] = __builtin_amdgcn_wmma_f32_16x16x32_bf16(
              false, a.v, false, bfr.v, (short)0, acc[n], false, false);
        }
      }
    }
  }

  // bias + store compact Z[pixel][64] fp32
#pragma unroll
  for (int n = 0; n < 4; ++n) {
    const float bias = b2[n * 16 + mrow];
#pragma unroll
    for (int e = 0; e < 8; ++e) {
      const int  row = e + hi * 8;
      const long p   = imgBase + (long)(y0 + wv) * HW + (x0 + row);
      Z[p * EX + n * 16 + mrow] = acc[n][e] + bias;
    }
  }
}

// ---------------------------------------------------------------------------
// commit: A[:, slice*64 : +64] = bf16(Z)   (strided fp32 -> bf16 merge)
// ---------------------------------------------------------------------------
__global__ __launch_bounds__(256) void commit_kernel(
    unsigned short* __restrict__ Abf, const float* __restrict__ Z, int slice)
{
  const long i   = (long)blockIdx.x * 256 + threadIdx.x;  // 1,048,576 float4
  const long pix = i >> 4;
  const int  c4  = (int)(i & 15);
  const float4 z = ((const float4*)Z)[i];
  uint2 pk;
  pk.x = pack2bf(z.x, z.y);
  pk.y = pack2bf(z.z, z.w);
  *(uint2*)&Abf[pix * CH + slice * EX + c4 * 4] = pk;
}

// ---------------------------------------------------------------------------
// step3: 256-step sequential channel recurrence in closed form.
// z_c = b + s_c + t_c ;  s_c = suffix sum of v_j*relu(a_j) ;
// t_c = P_c * prefix_{j<c}( v_j(b+s_j) / P_{j+1} ), P_c = prod_{j<c}(1+v_j).
// One wave per pixel; three wave scans (2 additive, 1 multiplicative).
// out = z + x (residual).
// ---------------------------------------------------------------------------
__global__ __launch_bounds__(256) void step3_kernel(
    const unsigned short* __restrict__ Abf, const float* __restrict__ x,
    const float* __restrict__ w3, const float* __restrict__ b3,
    float* __restrict__ out)
{
  const int  lane = threadIdx.x & 31;
  const long pix  = (long)blockIdx.x * 8 + (threadIdx.x >> 5);
  const float b   = b3[0];
  const int  cb   = lane * 8;

  U4H8 raw;
  raw.q = *(const uint4*)(Abf + pix * CH + cb);

  float v[8], a0[8];
#pragma unroll
  for (int k = 0; k < 8; ++k) {
    v[k]  = w3[cb + k];
    a0[k] = fmaxf(bf2f(raw.h[k]), 0.f);   // ReLU after step2, fused here
  }

  // scan 1: exclusive prefix of w_j = v_j * a0_j  -> suffix sums s_c
  float pre[8], loc = 0.f;
#pragma unroll
  for (int k = 0; k < 8; ++k) { pre[k] = loc; loc += v[k] * a0[k]; }
  float inc = loc;
#pragma unroll
  for (int off = 1; off < 32; off <<= 1) {
    float t = __shfl_up(inc, off, 32);
    if (lane >= off) inc += t;
  }
  const float exclLane = inc - loc;
  const float T = __shfl(inc, 31, 32);

  // scan 2 (multiplicative): P_c = prod_{j<c}(1 + v_j)
  float ppre[8], ploc = 1.f;
#pragma unroll
  for (int k = 0; k < 8; ++k) { ppre[k] = ploc; ploc *= (1.f + v[k]); }
  float pinc = ploc;
#pragma unroll
  for (int off = 1; off < 32; off <<= 1) {
    float t = __shfl_up(pinc, off, 32);
    if (lane >= off) pinc *= t;
  }
  const float pexcl = pinc / ploc;

  // scan 3: exclusive prefix of u_j = v_j(b+s_j)/P_{j+1}
  float s[8], Pj[8], upre[8], uloc = 0.f;
#pragma unroll
  for (int k = 0; k < 8; ++k) {
    s[k]  = T - (exclLane + pre[k]);
    Pj[k] = pexcl * ppre[k];
    const float beta = v[k] * (b + s[k]);
    const float u    = beta / (Pj[k] * (1.f + v[k]));
    upre[k] = uloc;
    uloc += u;
  }
  float uinc = uloc;
#pragma unroll
  for (int off = 1; off < 32; off <<= 1) {
    float t = __shfl_up(uinc, off, 32);
    if (lane >= off) uinc += t;
  }
  const float uexcl = uinc - uloc;

#pragma unroll
  for (int k = 0; k < 8; ++k) {
    const float tc = Pj[k] * (uexcl + upre[k]);
    const float z  = b + s[k] + tc;
    out[pix * CH + cb + k] = z + x[pix * CH + cb + k];
  }
}

// ---------------------------------------------------------------------------
extern "C" void kernel_launch(void* const* d_in, const int* in_sizes, int n_in,
                              void* d_out, int out_size, void* d_ws, size_t ws_size,
                              hipStream_t stream)
{
  (void)in_sizes; (void)n_in; (void)out_size; (void)ws_size;
  const float* x  = (const float*)d_in[0];
  const float* w1 = (const float*)d_in[1];
  const float* b1 = (const float*)d_in[2];
  const float* w2 = (const float*)d_in[3];
  const float* b2 = (const float*)d_in[4];
  const float* w3 = (const float*)d_in[5];
  const float* b3 = (const float*)d_in[6];
  float* out = (float*)d_out;

  char* ws = (char*)d_ws;
  unsigned short* Abf = (unsigned short*)(ws);                          // 32 MiB
  float*          Z   = (float*)(ws + ((size_t)32 << 20));              // 16 MiB
  unsigned short* w1T = (unsigned short*)(ws + ((size_t)48 << 20));     // 32 KiB
  unsigned short* w2T = (unsigned short*)(ws + ((size_t)48 << 20) + 65536); // 288 KiB
  unsigned short* zp  = (unsigned short*)(ws + ((size_t)48 << 20) + 524288); // 1 KiB zeros

  repack_kernel<<<256, 256, 0, stream>>>(w1, w2, w1T, w2T, zp);
  step1_kernel<<<NPIX / 64, 128, 0, stream>>>(x, w1T, b1, Abf);
  for (int i = 0; i < 4; ++i) {
    step2_kernel<<<dim3(HW / TW, HW / TH, 16), 128, 0, stream>>>(Abf, w2T, b2, Z, zp);
    commit_kernel<<<NPIX * EX / 4 / 256, 256, 0, stream>>>(Abf, Z, i);
  }
  step3_kernel<<<NPIX / 8, 256, 0, stream>>>(Abf, x, w3, b3, out);
}
